// KeopsAttentionProduct_62130996903971
// MI455X (gfx1250) — compile-verified, matrix operand order checked by reading, not run on hardware
//
#include <hip/hip_runtime.h>

#define T_SEQ   4096
#define DHEAD   64
#define WIN     256
#define NB      16      // T_SEQ / WIN
#define KW      512     // 2*WIN window
#define CH      32      // key chunk
#define KS_STR  72      // 64 + 8 pad: row pitch 144B = 36 dwords -> conflict-free b128
#define VT_STR  520     // 512 + 8 pad: col pitch 1040B = 260 dwords -> conflict-free b128

// exp(score/sqrt(64)) == exp2(score * 0.125 * log2(e))
#define SCALE_LOG2E 0.1803368801111204f

typedef __attribute__((ext_vector_type(16))) __bf16 v16bf;
typedef __attribute__((ext_vector_type(2)))  __bf16 v2bf;
typedef __attribute__((ext_vector_type(8)))  float  v8f;

union AFrag {
  v16bf v;
  unsigned short u[16];
  unsigned int   w[8];
  uint4 q[2];
};

__device__ __forceinline__ float fast_exp2(float x) {
#if __has_builtin(__builtin_amdgcn_exp2f)
  return __builtin_amdgcn_exp2f(x);     // v_exp_f32
#else
  return exp2f(x);
#endif
}

__device__ __forceinline__ unsigned short f2bf(float f) {
  unsigned int x = __float_as_uint(f);
  x += 0x7fffu + ((x >> 16) & 1u);          // round-to-nearest-even
  return (unsigned short)(x >> 16);
}

// pack two f32 -> 2xbf16 in one dword; let ISel pick v_cvt_pk_bf16_f32 if present
__device__ __forceinline__ unsigned int f2bf_pk(float lo, float hi) {
#if __has_builtin(__builtin_amdgcn_cvt_pk_bf16_f32)
  auto r = __builtin_amdgcn_cvt_pk_bf16_f32(lo, hi);
  unsigned int uo;
  __builtin_memcpy(&uo, &r, 4);
  return uo;
#else
  v2bf t;
  t[0] = (__bf16)lo;
  t[1] = (__bf16)hi;
  return __builtin_bit_cast(unsigned int, t);
#endif
}

// ---- DPP cross-lane (within 16-lane row) ----
template <int CTRL>
__device__ __forceinline__ float dpp_mov(float x) {
  return __builtin_bit_cast(float,
      __builtin_amdgcn_update_dpp(0, __builtin_bit_cast(int, x),
                                  CTRL, 0xf, 0xf, true));
}
// butterfly sum over the 16 lanes of a row (both halves independent)
__device__ __forceinline__ float red_sum16(float x) {
  x += dpp_mov<0xB1>(x);    // quad_perm [1,0,3,2]
  x += dpp_mov<0x4E>(x);    // quad_perm [2,3,0,1]
  x += dpp_mov<0x141>(x);   // row_half_mirror
  x += dpp_mov<0x140>(x);   // row_mirror
  return x;
}

// Build a 16-bit A-layout WMMA fragment from a row of f32 data (for Q).
__device__ __forceinline__ void frag_from_row(AFrag& f, const float* rowptr,
                                              int db0, int db1) {
  float4 a0 = *(const float4*)(rowptr + db0);
  float4 a1 = *(const float4*)(rowptr + db0 + 4);
  float4 b0 = *(const float4*)(rowptr + db1);
  float4 b1 = *(const float4*)(rowptr + db1 + 4);
  f.w[0] = f2bf_pk(a0.x, a0.y); f.w[1] = f2bf_pk(a0.z, a0.w);
  f.w[2] = f2bf_pk(a1.x, a1.y); f.w[3] = f2bf_pk(a1.z, a1.w);
  f.w[4] = f2bf_pk(b0.x, b0.y); f.w[5] = f2bf_pk(b0.z, b0.w);
  f.w[6] = f2bf_pk(b1.x, b1.y); f.w[7] = f2bf_pk(b1.z, b1.w);
}

extern "C" __global__ __launch_bounds__(512)
void local_attn_wmma(const float* __restrict__ Qg,
                     const float* __restrict__ Kg,
                     const float* __restrict__ Vg,
                     float* __restrict__ Og) {
  extern __shared__ char smem[];
  unsigned short* KS = (unsigned short*)smem;        // [512][KS_STR] bf16 keys, row-major
  unsigned short* VT = KS + KW * KS_STR;             // [64][VT_STR]  bf16 V^T, k-interleaved
  unsigned short* PB = VT + DHEAD * VT_STR;          // 16 waves x [16][32] bf16 P tiles

  const int blk  = blockIdx.x & (NB - 1);
  const int bh   = blockIdx.x >> 4;
  const int tid  = threadIdx.x;
  const int lane = tid & 31;
  const int wave = tid >> 5;
  const int hl   = lane >> 4;      // 16-lane half
  const int lr   = lane & 15;

  const long long base = (long long)bh * T_SEQ * DHEAD;
  const float* q = Qg + base;
  const float* k = Kg + base;
  const float* v = Vg + base;
  float* outp = Og + base;

  const int qbase  = blk * WIN;
  const int kstart = qbase - (WIN / 2);   // global key index of window slot 0

  // ---- stage K (bf16 rows, natural k) and V^T (bf16 cols, k-interleaved) ----
  #pragma unroll
  for (int it = 0; it < 16; ++it) {
    int fl  = tid + it * 512;       // float4 id over 512x64 tile
    int row = fl >> 4;              // window key 0..511
    int d0  = (fl & 15) << 2;
    int gk  = kstart + row;
    bool inb = ((unsigned)gk < (unsigned)T_SEQ);

    float4 kv = make_float4(0.f, 0.f, 0.f, 0.f);
    if (inb) kv = *(const float4*)(k + (long long)gk * DHEAD + d0);
    *(uint2*)(KS + row * KS_STR + d0) =
        make_uint2(f2bf_pk(kv.x, kv.y), f2bf_pk(kv.z, kv.w));

    float4 vv = make_float4(0.f, 0.f, 0.f, 0.f);
    if (inb) vv = *(const float4*)(v + (long long)gk * DHEAD + d0);
    // k-interleaved column index: iota(k) = 2*(k&15) + (k>>4) within each 32-chunk
    int kin  = row & 31;
    int colI = (row & ~31) + 2 * (kin & 15) + (kin >> 4);
    VT[(d0 + 0) * VT_STR + colI] = f2bf(vv.x);
    VT[(d0 + 1) * VT_STR + colI] = f2bf(vv.y);
    VT[(d0 + 2) * VT_STR + colI] = f2bf(vv.z);
    VT[(d0 + 3) * VT_STR + colI] = f2bf(vv.w);
  }
  __syncthreads();

  // ---- per-lane d-offset runs for 16-bit fragments (natural k order) ----
  const int db0_base = hl ? 8 : 0;
  const int dbl_base = hl ? 24 : 16;

  // ---- Q fragments (resident), rows qbase + wave*16 + lr ----
  const int qrow = qbase + wave * 16 + lr;
  AFrag qf[2];
  #pragma unroll
  for (int s = 0; s < 2; ++s)
    frag_from_row(qf[s], q + (long long)qrow * DHEAD, s * 32 + db0_base, s * 32 + dbl_base);

  // ---- accumulators & row-sum state (rows m = j + 8*hl of this wave's tile) ----
  v8f oacc[4];
  const v8f vzero = {0.f, 0.f, 0.f, 0.f, 0.f, 0.f, 0.f, 0.f};
  #pragma unroll
  for (int dt = 0; dt < 4; ++dt) oacc[dt] = vzero;
  float lrow[8];
  #pragma unroll
  for (int j = 0; j < 8; ++j) lrow[j] = 0.f;

  unsigned short* pb = PB + wave * (16 * 32);

  const int c_lo = wave >> 1;
  const int c_hi = (wave * 16 + 271) >> 5;   // last chunk with any in-band key

  for (int c = c_lo; c <= c_hi; ++c) {
    // ---- S = Q K^T for this 16x32 chunk (K tiles straight from LDS) ----
    v8f sf[2];
    sf[0] = vzero; sf[1] = vzero;
    #pragma unroll
    for (int nt = 0; nt < 2; ++nt) {
      const int g = c * CH + nt * 16 + lr;     // window key, always 0..511
      #pragma unroll
      for (int s = 0; s < 2; ++s) {
        AFrag kf;
        kf.q[0] = *(const uint4*)(KS + g * KS_STR + s * 32 + db0_base);
        kf.q[1] = *(const uint4*)(KS + g * KS_STR + s * 32 + dbl_base);
        sf[nt] = __builtin_amdgcn_wmma_f32_16x16x32_bf16(
            false, qf[s].v, false, kf.v, (short)0, sf[nt], false, false);
      }
    }

    // ---- mask + unnormalized exp (shift-free softmax), packed P to LDS ----
    #pragma unroll
    for (int j = 0; j < 8; ++j) {
      const int m  = j + 8 * hl;
      const int ql = wave * 16 + m;              // query index within 256-block
      float p[2];
      #pragma unroll
      for (int nt = 0; nt < 2; ++nt) {
        int kj = c * CH + nt * 16 + lr;          // window slot
        int gk = kstart + kj;
        bool valid = ((unsigned)(kj - ql) < (unsigned)WIN) &&
                     ((unsigned)gk < (unsigned)T_SEQ);
        float e = fast_exp2(sf[nt][j] * SCALE_LOG2E);
        p[nt] = valid ? e : 0.f;
      }
      lrow[j] += red_sum16(p[0] + p[1]);
      // k = lr and k = 16+lr interleave to adjacent slots 2*lr, 2*lr+1
      *(unsigned int*)(pb + m * 32 + 2 * lr) = f2bf_pk(p[0], p[1]);
    }

    // same-wave LDS store->load fence (also a compiler reordering barrier)
    asm volatile("s_wait_dscnt 0" ::: "memory");

    // ---- P fragment (A layout, k-interleaved: half maps to contiguous run) ----
    AFrag pf;
    pf.q[0] = *(const uint4*)(pb + lr * 32 + hl * 16);
    pf.q[1] = *(const uint4*)(pb + lr * 32 + hl * 16 + 8);

    // ---- O += P * V  (V^T k-interleaved columns, same permutation) ----
    #pragma unroll
    for (int dt = 0; dt < 4; ++dt) {
      AFrag vf;
      const int d = dt * 16 + lr;
      vf.q[0] = *(const uint4*)(VT + d * VT_STR + c * CH + hl * 16);
      vf.q[1] = *(const uint4*)(VT + d * VT_STR + c * CH + hl * 16 + 8);
      oacc[dt] = __builtin_amdgcn_wmma_f32_16x16x32_bf16(
          false, pf.v, false, vf.v, (short)0, oacc[dt], false, false);
    }
  }

  // ---- normalize and store ----
  float inv[8];
  #pragma unroll
  for (int j = 0; j < 8; ++j) inv[j] = 1.0f / lrow[j];
  #pragma unroll
  for (int dt = 0; dt < 4; ++dt) {
    const int d = dt * 16 + lr;
    #pragma unroll
    for (int j = 0; j < 8; ++j) {
      const int m = j + 8 * hl;
      outp[(long long)(qbase + wave * 16 + m) * DHEAD + d] = oacc[dt][j] * inv[j];
    }
  }
}

extern "C" void kernel_launch(void* const* d_in, const int* in_sizes, int n_in,
                              void* d_out, int out_size, void* d_ws, size_t ws_size,
                              hipStream_t stream) {
  (void)in_sizes; (void)n_in; (void)out_size; (void)d_ws; (void)ws_size;
  const float* Q = (const float*)d_in[0];
  const float* K = (const float*)d_in[1];
  const float* V = (const float*)d_in[2];
  float* O = (float*)d_out;

  const int BH = 2 * 12;
  dim3 grid(BH * NB);          // 384 tiles
  dim3 block(512);             // 16 waves (wave32)
  size_t shmem = (size_t)KW * KS_STR * 2      // K bf16 rows
               + (size_t)DHEAD * VT_STR * 2   // V^T bf16
               + (size_t)16 * 16 * 32 * 2;    // per-wave P buffers
  local_attn_wmma<<<grid, block, shmem, stream>>>(Q, K, V, O);
}
